// LiteLA_16552803958851
// MI455X (gfx1250) — compile-verified, matrix operand order checked by reading, not run on hardware
//
#include <hip/hip_runtime.h>

// ---------------------------------------------------------------------------
// LiteLA on MI455X (gfx1250, wave32):
//   GEMM1 (x @ w_qkv^T, relu on q/k, bf16 out)  -> v_wmma_f32_16x16x32_bf16
//   per-head linear attention (fp32 VALU, LDS, async global->LDS staging)
//   GEMM2 (attn @ w_proj^T + b)                 -> v_wmma_f32_16x16x32_bf16
// Compute-bound (275 GFLOP GEMM vs ~20us of HBM traffic at 23.3 TB/s), so the
// bf16 WMMA path with f32 accumulation is the right precision/throughput pick.
// ---------------------------------------------------------------------------

typedef __attribute__((ext_vector_type(16))) __bf16 v16bf;
typedef __attribute__((ext_vector_type(8)))  float  v8f;
typedef int v4i __attribute__((ext_vector_type(4)));
using gv4i = __attribute__((address_space(1))) v4i;   // global (AS1)
using sv4i = __attribute__((address_space(3))) v4i;   // LDS (AS3)

#define TILE_MN 128
#define TILE_K  64
#define LDSW    72   // ushorts per LDS row (64 + 8 pad) = 144B -> conflict-free b128

__device__ __forceinline__ unsigned short f2bf(float f) {
  unsigned int u = __builtin_bit_cast(unsigned int, f);
  u += 0x7FFFu + ((u >> 16) & 1u);              // round-to-nearest-even
  return (unsigned short)(u >> 16);
}
#if __has_builtin(__builtin_amdgcn_cvt_pk_bf16_f32)
__device__ __forceinline__ unsigned int pk2bf(float a, float b) {
  auto r = __builtin_amdgcn_cvt_pk_bf16_f32(a, b);
  return __builtin_bit_cast(unsigned int, r);
}
#else
__device__ __forceinline__ unsigned int pk2bf(float a, float b) {
  return (unsigned)f2bf(a) | ((unsigned)f2bf(b) << 16);
}
#endif
__device__ __forceinline__ float bf2f(unsigned short h) {
  return __builtin_bit_cast(float, ((unsigned int)h) << 16);
}
__device__ __forceinline__ float bflo(unsigned int u) { return __builtin_bit_cast(float, u << 16); }
__device__ __forceinline__ float bfhi(unsigned int u) { return __builtin_bit_cast(float, u & 0xFFFF0000u); }

// ISA 7.12.2: 16-bit A/B fragment = per-lane K-strip; two ds_load_b128 each.
__device__ __forceinline__ v16bf load_frag_lds(const unsigned short* rowbase, int koff) {
  union { uint4 q[2]; v16bf v; } u;
  u.q[0] = *reinterpret_cast<const uint4*>(rowbase + koff);
  u.q[1] = *reinterpret_cast<const uint4*>(rowbase + 16 + koff);
  return u.v;
}

// C = A(M x K) * B(Nt x K)^T ; A,B fp32 in global, converted to bf16 in staging.
// MODE 0: bf16 store, relu for gcol < relu_limit (qkv). MODE 1: f32 store + bias.
template<int MODE>
__global__ __launch_bounds__(256)
void wmma_gemm(const float* __restrict__ A, const float* __restrict__ Bw,
               void* __restrict__ Cout, int Ntot, int Ktot,
               const float* __restrict__ bias, int relu_limit)
{
  __shared__ __align__(16) unsigned short As[TILE_MN * LDSW];
  __shared__ __align__(16) unsigned short Bs[TILE_MN * LDSW];

  const int t    = threadIdx.x;
  const int wid  = t >> 5;             // 8 wave32 waves
  const int lane = t & 31;
  const int wm   = wid & 3;            // 4 wave rows -> 32 output rows each
  const int wn   = wid >> 2;           // 2 wave cols -> 64 output cols each
  const int rh   = lane & 15;
  const int koff = (lane >> 4) << 3;   // 0 or 8 (half-wave K offset)

  const size_t rowA0 = (size_t)blockIdx.y * TILE_MN;
  const size_t rowB0 = (size_t)blockIdx.x * TILE_MN;

  v8f acc[2][4] = {};

  for (int kt = 0; kt < Ktot; kt += TILE_K) {
    // ---- stage A & B 128x64 tiles: global f32 -> bf16 -> LDS ----
    #pragma unroll
    for (int i = 0; i < 8; i++) {
      const int f4  = t + i * 256;     // 2048 float4's per matrix tile
      const int row = f4 >> 4;         // 0..127
      const int c4  = f4 & 15;         // float4 column within 64-wide K slab
      const float4 a = *reinterpret_cast<const float4*>(A  + (rowA0 + row) * Ktot + kt + c4 * 4);
      const float4 b = *reinterpret_cast<const float4*>(Bw + (rowB0 + row) * Ktot + kt + c4 * 4);
      uint2 pa, pb;
      pa.x = pk2bf(a.x, a.y);  pa.y = pk2bf(a.z, a.w);
      pb.x = pk2bf(b.x, b.y);  pb.y = pk2bf(b.z, b.w);
      *reinterpret_cast<uint2*>(&As[row * LDSW + c4 * 4]) = pa;
      *reinterpret_cast<uint2*>(&Bs[row * LDSW + c4 * 4]) = pb;
      if ((kt + TILE_K < Ktot) && !(c4 & 1)) {   // global_prefetch next K slab
        __builtin_prefetch(A  + (rowA0 + row) * Ktot + kt + TILE_K + c4 * 4, 0, 1);
        __builtin_prefetch(Bw + (rowB0 + row) * Ktot + kt + TILE_K + c4 * 4, 0, 1);
      }
    }
    __syncthreads();

    // ---- 16 WMMAs per wave per K-tile (2 sub-K x 2 M-sub x 4 N-sub) ----
    #pragma unroll
    for (int kk = 0; kk < 2; kk++) {
      v16bf af[2], bfr[4];
      #pragma unroll
      for (int s = 0; s < 2; s++)
        af[s] = load_frag_lds(&As[(wm * 32 + s * 16 + rh) * LDSW + kk * 32], koff);
      #pragma unroll
      for (int u = 0; u < 4; u++)
        bfr[u] = load_frag_lds(&Bs[(wn * 64 + u * 16 + rh) * LDSW + kk * 32], koff);
      #pragma unroll
      for (int s = 0; s < 2; s++)
        #pragma unroll
        for (int u = 0; u < 4; u++)
          acc[s][u] = __builtin_amdgcn_wmma_f32_16x16x32_bf16(
              false, af[s], false, bfr[u], (short)0, acc[s][u], false, false);
    }
    __syncthreads();
  }

  // ---- epilogue: C layout: lane -> N, vgpr r -> M (+8 for hi half-wave) ----
  const int mlo = (lane < 16) ? 0 : 8;
  const int nl  = lane & 15;
  #pragma unroll
  for (int s = 0; s < 2; s++) {
    #pragma unroll
    for (int u = 0; u < 4; u++) {
      #pragma unroll
      for (int r = 0; r < 8; r++) {
        const size_t grow = rowA0 + wm * 32 + s * 16 + r + mlo;
        const int    gcol = (int)rowB0 + wn * 64 + u * 16 + nl;
        float v = acc[s][u][r];
        if (MODE == 0) {
          if (gcol < relu_limit) v = fmaxf(v, 0.0f);
          reinterpret_cast<unsigned short*>(Cout)[grow * (size_t)Ntot + gcol] = f2bf(v);
        } else {
          reinterpret_cast<float*>(Cout)[grow * (size_t)Ntot + gcol] = v + bias[gcol];
        }
      }
    }
  }
}

// ---------------------------------------------------------------------------
// Per-head linear attention; one block per (b, head). qkv bf16 [B*N][3*C],
// relu already folded into q/k. k/v tiles staged raw (bf16) into LDS via
// async global->LDS copies when available (ASYNCcnt-tracked), else uint4 copy.
//   vk[e][dd] = sum_n v_pad[e][n] * k[dd][n]   (33x32, K=4096)
//   out[e][n] = sum_dd vk[e][dd] * q[dd][n] ; attn = out[:32]/(out[32]+eps)
// ---------------------------------------------------------------------------
#define ATN  128
#define ALDW 40   // ushorts per LDS row (32 + 8 pad), 80B: 16B-aligned b128 slots

#if __has_builtin(__builtin_amdgcn_global_load_async_to_lds_b128)
#define HAVE_ASYNC_LDS 1
#endif

__global__ __launch_bounds__(256)
void litela_attn(const unsigned short* __restrict__ qkv, float* __restrict__ attn)
{
  const int N = 4096, C = 2048;
  const int bh = blockIdx.x;
  const int b  = bh >> 6;       // 64 heads
  const int hh = bh & 63;

  __shared__ __align__(16) unsigned short kraw[ATN * ALDW];  // [n][dd] bf16
  __shared__ __align__(16) unsigned short vraw[ATN * ALDW];
  __shared__ float vk[33][32];

  const int t = threadIdx.x;
  float acc[5] = {0.f, 0.f, 0.f, 0.f, 0.f};

  for (int n0 = 0; n0 < N; n0 += ATN) {
    __syncthreads();
    #pragma unroll
    for (int i = 0; i < 2; i++) {
      const int id  = t + i * 256;   // 512 x 16B chunks per tile
      const int nl  = id >> 2;
      const int dd0 = (id & 3) * 8;
      const size_t rowK = ((size_t)(b * N + n0 + nl) * 3 + 1) * C + hh * 32 + dd0;
      const size_t rowV = ((size_t)(b * N + n0 + nl) * 3 + 2) * C + hh * 32 + dd0;
#ifdef HAVE_ASYNC_LDS
      __builtin_amdgcn_global_load_async_to_lds_b128(
          (gv4i*)(qkv + rowK), (sv4i*)&kraw[nl * ALDW + dd0], 0, 0);
      __builtin_amdgcn_global_load_async_to_lds_b128(
          (gv4i*)(qkv + rowV), (sv4i*)&vraw[nl * ALDW + dd0], 0, 0);
#else
      *reinterpret_cast<uint4*>(&kraw[nl * ALDW + dd0]) =
          *reinterpret_cast<const uint4*>(qkv + rowK);
      *reinterpret_cast<uint4*>(&vraw[nl * ALDW + dd0]) =
          *reinterpret_cast<const uint4*>(qkv + rowV);
#endif
    }
#ifdef HAVE_ASYNC_LDS
    asm volatile("s_wait_asynccnt 0" ::: "memory");
#endif
    __syncthreads();
    // each thread owns up to 5 (e,dd) pairs of the 33x32 vk accumulator
    #pragma unroll
    for (int p = 0; p < 5; p++) {
      const int pair = t + p * 256;
      if (pair < 1056) {
        const int e  = pair >> 5;
        const int dd = pair & 31;
        float s = 0.f;
        if (e == 32) {                       // PAD_VAL == 1.0 row
          for (int n = 0; n < ATN; n++) s += bf2f(kraw[n * ALDW + dd]);
        } else {
          for (int n = 0; n < ATN; n++)
            s += bf2f(vraw[n * ALDW + e]) * bf2f(kraw[n * ALDW + dd]);
        }
        acc[p] += s;
      }
    }
  }
  __syncthreads();
  #pragma unroll
  for (int p = 0; p < 5; p++) {
    const int pair = t + p * 256;
    if (pair < 1056) vk[pair >> 5][pair & 31] = acc[p];
  }
  __syncthreads();

  for (int n = t; n < N; n += 256) {
    const size_t rowQ = ((size_t)(b * N + n) * 3 + 0) * C + hh * 32;
    float q[32];
    #pragma unroll
    for (int j4 = 0; j4 < 4; j4++) {
      const uint4 uq = *reinterpret_cast<const uint4*>(qkv + rowQ + j4 * 8);
      const unsigned int qu[4] = {uq.x, uq.y, uq.z, uq.w};
      #pragma unroll
      for (int j = 0; j < 4; j++) {
        q[j4*8 + 2*j + 0] = bflo(qu[j]);
        q[j4*8 + 2*j + 1] = bfhi(qu[j]);
      }
    }
    float den = 0.f;
    #pragma unroll
    for (int dd = 0; dd < 32; dd++) den += vk[32][dd] * q[dd];
    const float w = 1.0f / (den + 1e-15f);
    float* dst = attn + (size_t)(b * N + n) * C + hh * 32;
    #pragma unroll
    for (int e = 0; e < 32; e++) {
      float o = 0.f;
      #pragma unroll
      for (int dd = 0; dd < 32; dd++) o += vk[e][dd] * q[dd];
      dst[e] = o * w;
    }
  }
}

// ---------------------------------------------------------------------------
extern "C" void kernel_launch(void* const* d_in, const int* in_sizes, int n_in,
                              void* d_out, int out_size, void* d_ws, size_t ws_size,
                              hipStream_t stream) {
  (void)in_sizes; (void)n_in; (void)out_size; (void)ws_size;
  const float* x      = (const float*)d_in[0];   // [2,4096,2048]
  const float* w_qkv  = (const float*)d_in[1];   // [6144,2048]
  const float* w_proj = (const float*)d_in[2];   // [2048,2048]
  const float* b_proj = (const float*)d_in[3];   // [2048]
  float* out = (float*)d_out;                    // [2,4096,2048]

  const int BN = 2 * 4096;     // 8192 rows
  const int C  = 2048;
  const int N3 = 3 * C;        // 6144

  // workspace: qkv bf16 [8192][6144] (100.7MB) then attn f32 [8192][2048] (67MB)
  char* ws = (char*)d_ws;
  unsigned short* qkv_bf = (unsigned short*)ws;
  float* attn = (float*)(ws + (size_t)BN * N3 * sizeof(unsigned short));

  dim3 blk(256);
  // GEMM1: qkv = x @ w_qkv^T, relu on cols < 4096 (q,k), bf16 out
  wmma_gemm<0><<<dim3(N3 / TILE_MN, BN / TILE_MN), blk, 0, stream>>>(
      x, w_qkv, (void*)qkv_bf, N3, C, nullptr, 2 * C);
  // per-head linear attention
  litela_attn<<<dim3(128), blk, 0, stream>>>(qkv_bf, attn);
  // GEMM2: out = attn @ w_proj^T + b_proj, fp32 out
  wmma_gemm<1><<<dim3(C / TILE_MN, BN / TILE_MN), blk, 0, stream>>>(
      attn, w_proj, (void*)out, C, C, b_proj, 0);
}
